// AttentionJES_56057913147404
// MI455X (gfx1250) — matrix operand order repaired and checked by reading, not compiled, not run on hardware
//
#include <hip/hip_runtime.h>
#include <hip/hip_bf16.h>
#include <math.h>

// ---------------------------------------------------------------------------
// Attention: out = softmax(mask(Q K^T / sqrt(D))) V,  Q=xWq, K=xWk, V=xWv
// B=8, S=2048, D=A=1024.  d_out = [output | Q | K | V] (f32).
// bf16 WMMA (f32 accum); scores (128MB, L2-resident) in d_ws; softmax
// rewrites P in place as bf16; PV GEMM consumes bf16 A directly.
// Tiles staged to LDS with global_load_async_to_lds_b128 (ASYNCcnt, inline
// asm), double-buffered; k-major B operands pipelined through VGPRs.
// ---------------------------------------------------------------------------

typedef __attribute__((ext_vector_type(16))) __bf16 v16bf;
typedef __attribute__((ext_vector_type(8)))  __bf16 v8bf;
typedef __attribute__((ext_vector_type(8)))  float  v8f;
typedef __attribute__((ext_vector_type(4)))  float  v4f;

#define TILE_M 128
#define TILE_N 128
#define TILE_K 32
#define NTHR   256   // 8 waves (wave32)

// Async global->LDS copy, 16 bytes per lane. vdst VGPR = 32-bit wave-relative
// LDS byte offset (low 32 bits of the generic pointer), vaddr = 64-bit global
// address, saddr = off (NULL). Tracked by ASYNCcnt.
__device__ __forceinline__ void async_copy_b128(const void* g, void* l)
{
#if defined(__AMDGCN__)
    asm volatile("global_load_async_to_lds_b128 %0, %1, off"
                 :: "v"((unsigned)(unsigned long long)l),
                    "v"((unsigned long long)g)
                 : "memory");
#else
    *(v4f*)l = *(const v4f*)g;
#endif
}

template<int N>
__device__ __forceinline__ void wait_async()
{
#if defined(__AMDGCN__)
#if __has_builtin(__builtin_amdgcn_s_wait_asynccnt)
    __builtin_amdgcn_s_wait_asynccnt((unsigned short)N);
#else
    asm volatile("s_wait_asynccnt %0" :: "i"(N));
#endif
#endif
}

template<bool C, class X, class Y> struct CondT { using T = X; };
template<class X, class Y> struct CondT<false, X, Y> { using T = Y; };

// AMODE: 0 = A is f32 (async-staged as f32, cvt at fragment build)
//        1 = A is bf16 (async-staged directly)
// BMODE: 0 = B given [K,N] row-major (VGPR-pipelined transpose to LDS bf16)
//        1 = B given [N,K] row-major (async-staged as f32)
// EPI  : 0 = plain store, 1 = scale + optional causal mask (scores)
template<int AMODE, int BMODE, int EPI>
__global__ __launch_bounds__(NTHR)
void gemm_wmma_kernel(const void* __restrict__ Av, long lda, long strideA,
                      const void* __restrict__ Bv, long ldb, long strideB,
                      float* __restrict__ C, long ldc, long strideC,
                      int Kd, float scale, const int* __restrict__ maskFlag)
{
    constexpr int AROW = (AMODE == 1) ? 40 : 36;  // padded rows (bank-spread)
    constexpr int BROW = (BMODE == 1) ? 36 : 40;
    using ATy = typename CondT<AMODE == 1, __bf16, float>::T;
    using BTy = typename CondT<BMODE == 1, float, __bf16>::T;
    __shared__ ATy As[2][TILE_M][AROW];
    __shared__ BTy Bs[2][TILE_N][BROW];

    constexpr int A_OPS = (AMODE == 1) ? 2 : 4;   // async instrs / thread / stage
    constexpr int B_OPS = (BMODE == 1) ? 4 : 0;
    constexpr int S_OPS = A_OPS + B_OPS;

    const int tid  = threadIdx.x;
    const int wave = tid >> 5;
    const int lane = tid & 31;
    const int hi   = lane >> 4;
    const int l16  = lane & 15;

    const int  bz    = blockIdx.z;
    const long mBase = (long)blockIdx.y * TILE_M;
    const long nBase = (long)blockIdx.x * TILE_N;

    const int waveM = (wave >> 1) * 32;   // 0,32,64,96
    const int waveN = (wave & 1) * 64;    // 0,64

    v8f acc[2][4];
#pragma unroll
    for (int mt = 0; mt < 2; ++mt)
#pragma unroll
        for (int nt = 0; nt < 4; ++nt)
            acc[mt][nt] = (v8f){};

    const bool doMask      = (EPI == 1) && (maskFlag != nullptr) && (maskFlag[0] != 0);
    const bool fullyMasked = doMask && (nBase > mBase + (TILE_M - 1));

    const float*  Af = (const float*)Av  + strideA * bz;   // AMODE==0
    const __bf16* Ah = (const __bf16*)Av + strideA * bz;   // AMODE==1
    const float*  Bf = (const float*)Bv  + strideB * bz;

    // ---- staging helpers -------------------------------------------------
    auto stageA = [&](int k0, int buf) {
        if constexpr (AMODE == 0) {
#pragma unroll
            for (int i = 0; i < 4; ++i) {           // 1024 x 16B chunks
                const int c   = tid + i * NTHR;
                const int row = c >> 3;
                const int sub = c & 7;
                async_copy_b128(Af + (mBase + row) * lda + k0 + sub * 4,
                                (void*)&As[buf][row][sub * 4]);
            }
        } else {
#pragma unroll
            for (int i = 0; i < 2; ++i) {           // 512 x 16B chunks (bf16)
                const int c   = tid + i * NTHR;
                const int row = c >> 2;
                const int sub = c & 3;
                async_copy_b128(Ah + (mBase + row) * lda + k0 + sub * 8,
                                (void*)&As[buf][row][sub * 8]);
            }
        }
    };
    auto stageB_async = [&](int k0, int buf) {       // BMODE==1: B is [N,K]
        if constexpr (BMODE == 1) {
#pragma unroll
            for (int i = 0; i < 4; ++i) {
                const int c   = tid + i * NTHR;
                const int row = c >> 3;
                const int sub = c & 7;
                async_copy_b128(Bf + (nBase + row) * ldb + k0 + sub * 4,
                                (void*)&Bs[buf][row][sub * 4]);
            }
        }
    };
    v4f breg[4];                                     // BMODE==0 pipeline regs
    auto loadB = [&](int k0) {                       // issue early, no wait
        if constexpr (BMODE == 0) {
            const int kr = tid >> 3;                 // 0..31
            const int nc = (tid & 7) * 16;           // 0..112
            const float* p = Bf + (long)(k0 + kr) * ldb + nBase + nc;
            breg[0] = *(const v4f*)(p + 0);
            breg[1] = *(const v4f*)(p + 4);
            breg[2] = *(const v4f*)(p + 8);
            breg[3] = *(const v4f*)(p + 12);
        }
    };
    auto storeB = [&](int buf) {                     // cvt + transpose to LDS
        if constexpr (BMODE == 0) {
            const int kr = tid >> 3;
            const int nc = (tid & 7) * 16;
#pragma unroll
            for (int i = 0; i < 4; ++i)
#pragma unroll
                for (int j = 0; j < 4; ++j)
                    Bs[buf][nc + i * 4 + j][kr] = (__bf16)breg[i][j];
        }
    };

    // ---- fragment build + WMMA ------------------------------------------
    auto compute = [&](int buf) {
        v16bf afrag[2];
#pragma unroll
        for (int mt = 0; mt < 2; ++mt) {
            const int row = waveM + mt * 16 + l16;
            if constexpr (AMODE == 0) {
                const float* ar = (const float*)&As[buf][row][hi * 8];
                v4f a0 = *(const v4f*)(ar + 0);
                v4f a1 = *(const v4f*)(ar + 4);
                v4f a2 = *(const v4f*)(ar + 16);
                v4f a3 = *(const v4f*)(ar + 20);
                v16bf a;
#pragma unroll
                for (int j = 0; j < 4; ++j) {
                    a[j]      = (__bf16)a0[j];
                    a[4 + j]  = (__bf16)a1[j];
                    a[8 + j]  = (__bf16)a2[j];
                    a[12 + j] = (__bf16)a3[j];
                }
                afrag[mt] = a;
            } else {
                v8bf* ap = (v8bf*)&afrag[mt];
                ap[0] = *(const v8bf*)&As[buf][row][hi * 8];
                ap[1] = *(const v8bf*)&As[buf][row][16 + hi * 8];
            }
        }
        v16bf bfrag[4];
#pragma unroll
        for (int nt = 0; nt < 4; ++nt) {
            const int col = waveN + nt * 16 + l16;
            if constexpr (BMODE == 1) {
                const float* br = (const float*)&Bs[buf][col][hi * 8];
                v4f b0 = *(const v4f*)(br + 0);
                v4f b1 = *(const v4f*)(br + 4);
                v4f b2 = *(const v4f*)(br + 16);
                v4f b3 = *(const v4f*)(br + 20);
                v16bf b;
#pragma unroll
                for (int j = 0; j < 4; ++j) {
                    b[j]      = (__bf16)b0[j];
                    b[4 + j]  = (__bf16)b1[j];
                    b[8 + j]  = (__bf16)b2[j];
                    b[12 + j] = (__bf16)b3[j];
                }
                bfrag[nt] = b;
            } else {
                v8bf* bp = (v8bf*)&bfrag[nt];
                bp[0] = *(const v8bf*)&Bs[buf][col][hi * 8];
                bp[1] = *(const v8bf*)&Bs[buf][col][16 + hi * 8];
            }
        }
#pragma unroll
        for (int mt = 0; mt < 2; ++mt)
#pragma unroll
            for (int nt = 0; nt < 4; ++nt)
                acc[mt][nt] = __builtin_amdgcn_wmma_f32_16x16x32_bf16(
                    false, afrag[mt], false, bfrag[nt],
                    (short)0, acc[mt][nt], false, false);
    };

    // ---- double-buffered pipeline ---------------------------------------
    if (!fullyMasked) {
        const int Kt = Kd / TILE_K;
        stageA(0, 0);
        if constexpr (BMODE == 1) {
            stageB_async(0, 0);
        } else {
            loadB(0);
            storeB(0);
        }
        for (int k = 0; k < Kt; ++k) {
            const int  cur  = k & 1;
            const bool more = (k + 1 < Kt);
            if (more) {
                stageA((k + 1) * TILE_K, cur ^ 1);
                if constexpr (BMODE == 1) stageB_async((k + 1) * TILE_K, cur ^ 1);
                else                      loadB((k + 1) * TILE_K);
            }
            if (more) wait_async<S_OPS>();   // stage k done (in-order)
            else      wait_async<0>();
            __syncthreads();
            compute(cur);
            if (BMODE == 0 && more) storeB(cur ^ 1);
            __syncthreads();
        }
    }

    // ---- epilogue --------------------------------------------------------
    float* Cb = C + strideC * bz;
#pragma unroll
    for (int mt = 0; mt < 2; ++mt) {
#pragma unroll
        for (int nt = 0; nt < 4; ++nt) {
            const long col = nBase + waveN + nt * 16 + l16;
#pragma unroll
            for (int r = 0; r < 8; ++r) {
                const long row = mBase + waveM + mt * 16 + r + hi * 8;
                float v = acc[mt][nt][r];
                if (EPI == 1) {
                    v *= scale;
                    if (doMask && col > row) v = -__builtin_inff();
                }
                Cb[row * ldc + col] = v;
            }
        }
    }
}

// ---------------------------------------------------------------------------
// Row softmax over S=2048 f32 scores; writes P back IN PLACE as bf16.
// One block (256 thr) per (q, b) row.
// ---------------------------------------------------------------------------
__global__ __launch_bounds__(256)
void softmax_rows_kernel(float* __restrict__ scores, int S,
                         long rowStride, long batchStride)
{
    const int q = blockIdx.x;
    const int b = blockIdx.y;
    float* row = scores + (long)b * batchStride + (long)q * rowStride;

    const int tid  = threadIdx.x;
    const int wave = tid >> 5;
    const int lane = tid & 31;

    float v[8];
#pragma unroll
    for (int i = 0; i < 8; ++i)
        v[i] = row[tid + i * 256];

    float m = -__builtin_inff();
#pragma unroll
    for (int i = 0; i < 8; ++i) m = fmaxf(m, v[i]);
#pragma unroll
    for (int off = 16; off > 0; off >>= 1)
        m = fmaxf(m, __shfl_xor(m, off, 32));

    __shared__ float red[8];
    if (lane == 0) red[wave] = m;
    __syncthreads();
    if (tid == 0) {
        float mm = red[0];
#pragma unroll
        for (int w = 1; w < 8; ++w) mm = fmaxf(mm, red[w]);
        red[0] = mm;
    }
    __syncthreads();
    m = red[0];
    __syncthreads();

    float s = 0.0f;
#pragma unroll
    for (int i = 0; i < 8; ++i) {
        v[i] = __expf(v[i] - m);          // exp(-inf - m) == 0 handles mask
        s += v[i];
    }
#pragma unroll
    for (int off = 16; off > 0; off >>= 1)
        s += __shfl_xor(s, off, 32);
    if (lane == 0) red[wave] = s;
    __syncthreads();
    if (tid == 0) {
        float ss = red[0];
#pragma unroll
        for (int w = 1; w < 8; ++w) ss += red[w];
        red[0] = ss;
    }
    __syncthreads();
    const float inv = 1.0f / red[0];

    __bf16* prow = (__bf16*)row;          // all reads done before 1st barrier
#pragma unroll
    for (int i = 0; i < 8; ++i)
        prow[tid + i * 256] = (__bf16)(v[i] * inv);
}

// ---------------------------------------------------------------------------
extern "C" void kernel_launch(void* const* d_in, const int* in_sizes, int n_in,
                              void* d_out, int out_size, void* d_ws, size_t ws_size,
                              hipStream_t stream)
{
    constexpr int  B = 8, S = 2048, D = 1024, A = 1024;
    constexpr long BSA = (long)B * S * A;

    const float* x        = (const float*)d_in[0];
    const float* Wq       = (const float*)d_in[1];
    const float* Wk       = (const float*)d_in[2];
    const float* Wv       = (const float*)d_in[3];
    const int*   maskFlag = (const int*)d_in[4];

    float* out = (float*)d_out;
    float* Q   = out + BSA;
    float* K   = out + 2 * BSA;
    float* V   = out + 3 * BSA;

    float* scores = (float*)d_ws;          // B*S*S f32 = 128 MB (L2-resident)

    const dim3  blk(NTHR);
    const float scale = 1.0f / sqrtf((float)D);

    // 1) Q/K/V projections: [B*S,D] x [D,A]
    {
        dim3 g(A / TILE_N, (B * S) / TILE_M, 1);
        gemm_wmma_kernel<0, 0, 0><<<g, blk, 0, stream>>>(
            x, D, 0, Wq, A, 0, Q, A, 0, D, 1.0f, nullptr);
        gemm_wmma_kernel<0, 0, 0><<<g, blk, 0, stream>>>(
            x, D, 0, Wk, A, 0, K, A, 0, D, 1.0f, nullptr);
        gemm_wmma_kernel<0, 0, 0><<<g, blk, 0, stream>>>(
            x, D, 0, Wv, A, 0, V, A, 0, D, 1.0f, nullptr);
    }

    // 2) scores = Q K^T * scale, causal mask (K is the n-major B operand)
    {
        dim3 g(S / TILE_N, S / TILE_M, B);
        gemm_wmma_kernel<0, 1, 1><<<g, blk, 0, stream>>>(
            Q, A, (long)S * A, K, A, (long)S * A,
            scores, S, (long)S * S, A, scale, maskFlag);
    }

    // 3) row softmax, P written in place as bf16
    softmax_rows_kernel<<<dim3(S, B), dim3(256), 0, stream>>>(
        scores, S, S, (long)S * S);

    // 4) out = P V  (A operand bf16, row stride 2*S bf16 elements)
    {
        dim3 g(A / TILE_N, S / TILE_M, B);
        gemm_wmma_kernel<1, 0, 0><<<g, blk, 0, stream>>>(
            scores, 2 * S, (long)S * 2 * S, V, A, (long)S * A,
            out, A, (long)S * A, S, 1.0f, nullptr);
    }
}
